// RelativeMultiHeadAttention_26182120636444
// MI455X (gfx1250) — matrix operand order repaired and checked by reading, not compiled, not run on hardware
//
#include <hip/hip_runtime.h>
#include <hip/hip_bf16.h>

typedef __attribute__((ext_vector_type(16))) __bf16 v16bf;
typedef __attribute__((ext_vector_type(8)))  __bf16 v8bf;
typedef __attribute__((ext_vector_type(8)))  float  v8f;

#define B_   2
#define S_   2048
#define E_   512
#define H_   8
#define HD_  64
#define P_   34
#define PP_  48
#define NEG_BIG (-1e30f)

// ---------------- helpers ----------------

__device__ __forceinline__ v8f v8f_zero() {
  v8f z;
#pragma unroll
  for (int i = 0; i < 8; ++i) z[i] = 0.0f;
  return z;
}

__device__ __forceinline__ v8f wmma_bf16(v16bf a, v16bf b, v8f c) {
  // D = A(16x32 bf16) * B(32x16 bf16) + C(f32)
  return __builtin_amdgcn_wmma_f32_16x16x32_bf16(false, a, false, b, (short)0, c,
                                                 false, false);
}

// A-operand fragment (16x32 bf16). rp = pointer to this lane's matrix row
// (row m = lane%16). half = lane/16. K runs: [k0+8h, +8) and [k0+16+8h, +8).
__device__ __forceinline__ v16bf ldfragA(const __bf16* rp, int k0, int half) {
  v8bf lo = *(const v8bf*)(rp + k0 + 8 * half);
  v8bf hi = *(const v8bf*)(rp + k0 + 16 + 8 * half);
  return __builtin_shufflevector(lo, hi, 0, 1, 2, 3, 4, 5, 6, 7,
                                 8, 9, 10, 11, 12, 13, 14, 15);
}

// B-operand fragment (32x16 bf16). rp = pointer to the memory row that holds
// this lane's column (col = lane%16) with contiguous K. K runs: [k0+8h, +8) and [k0+16+8h, +8).
__device__ __forceinline__ v16bf ldfragB(const __bf16* rp, int k0, int half) {
  v8bf lo = *(const v8bf*)(rp + k0 + 8 * half);
  v8bf hi = *(const v8bf*)(rp + k0 + 16 + 8 * half);
  return __builtin_shufflevector(lo, hi, 0, 1, 2, 3, 4, 5, 6, 7,
                                 8, 9, 10, 11, 12, 13, 14, 15);
}

// A-operand fragment with on-the-fly f32 -> bf16 conversion.
__device__ __forceinline__ v16bf ldfragA_f32(const float* rp, int k0, int half) {
  const float4 x0 = *(const float4*)(rp + k0 + 8 * half);
  const float4 x1 = *(const float4*)(rp + k0 + 8 * half + 4);
  const float4 y0 = *(const float4*)(rp + k0 + 16 + 8 * half);
  const float4 y1 = *(const float4*)(rp + k0 + 16 + 8 * half + 4);
  v16bf a;
  a[0]  = (__bf16)x0.x; a[1]  = (__bf16)x0.y; a[2]  = (__bf16)x0.z; a[3]  = (__bf16)x0.w;
  a[4]  = (__bf16)x1.x; a[5]  = (__bf16)x1.y; a[6]  = (__bf16)x1.z; a[7]  = (__bf16)x1.w;
  a[8]  = (__bf16)y0.x; a[9]  = (__bf16)y0.y; a[10] = (__bf16)y0.z; a[11] = (__bf16)y0.w;
  a[12] = (__bf16)y1.x; a[13] = (__bf16)y1.y; a[14] = (__bf16)y1.z; a[15] = (__bf16)y1.w;
  return a;
}

// reductions across the 16 lanes that hold one C-matrix row (wave32)
__device__ __forceinline__ float rmax16(float v) {
#pragma unroll
  for (int off = 1; off < 16; off <<= 1) v = fmaxf(v, __shfl_xor(v, off, 32));
  return v;
}
__device__ __forceinline__ float rsum16(float v) {
#pragma unroll
  for (int off = 1; off < 16; off <<= 1) v += __shfl_xor(v, off, 32);
  return v;
}

// ---------------- kernel 1: convert weights / rel table to bf16 ----------------
__global__ __launch_bounds__(256) void cvt_kernel(const float* Wq, const float* Wk,
                                                  const float* Wv, const float* Wo,
                                                  const float* rel, __bf16* Wbf,
                                                  __bf16* Rh) {
  const int idx = blockIdx.x * 256 + threadIdx.x;
  const int WTOT = 4 * E_ * E_;
  if (idx < WTOT) {
    const int w = idx >> 18;          // which matrix (E_*E_ = 2^18)
    const int r = idx & (E_ * E_ - 1);
    const float* src = (w == 0) ? Wq : (w == 1) ? Wk : (w == 2) ? Wv : Wo;
    Wbf[idx] = (__bf16)src[r];
  } else {
    const int j = idx - WTOT;
    if (j < H_ * PP_ * HD_) {
      const int h = j / (PP_ * HD_);
      const int rm = j % (PP_ * HD_);
      const int p = rm / HD_;
      const int d = rm % HD_;
      Rh[j] = (p < P_) ? (__bf16)rel[p * E_ + d * H_ + h] : (__bf16)0.0f;
    }
  }
}

// ---------------- kernel 2: projection GEMM Y = X * W^T, head-split store -----
// Ping-pong software pipeline (2x unrolled): buffers alternate roles, so there
// are no cross-iteration register copies (and no WAR hazard nops).
// mode 0: dst[b][h][s][d]   (Qp / Kp)
// mode 1: dst[b][h][d][s]   (V transposed for the P*V B-operand)
__global__ __launch_bounds__(256) void proj_kernel(const float* __restrict__ X,
                                                   const __bf16* __restrict__ W,
                                                   __bf16* __restrict__ dst, int mode) {
  const int lane = threadIdx.x & 31;
  const int wave = threadIdx.x >> 5;
  const int l16 = lane & 15;
  const int half = lane >> 4;
  const int n0 = blockIdx.x * 16;   // token rows
  const int c0 = wave * 64;         // output channels

  const float* xrow = X + (size_t)(n0 + l16) * E_;
  const __bf16* wrow[4];
#pragma unroll
  for (int sub = 0; sub < 4; ++sub)
    wrow[sub] = W + (size_t)(c0 + sub * 16 + l16) * E_;

  v8f acc[4];
#pragma unroll
  for (int i = 0; i < 4; ++i) acc[i] = v8f_zero();

  // prologue: tile 0 into buffer 0
  v16bf a0 = ldfragA_f32(xrow, 0, half);
  v16bf b0[4];
#pragma unroll
  for (int sub = 0; sub < 4; ++sub) b0[sub] = ldfragB(wrow[sub], 0, half);

#pragma unroll 1
  for (int k0 = 0; k0 < E_ - 64; k0 += 64) {
    // phase 0: prefetch k0+32 into buffer 1, compute with buffer 0
    v16bf a1 = ldfragA_f32(xrow, k0 + 32, half);
    v16bf b1[4];
#pragma unroll
    for (int sub = 0; sub < 4; ++sub) b1[sub] = ldfragB(wrow[sub], k0 + 32, half);
#pragma unroll
    for (int sub = 0; sub < 4; ++sub) acc[sub] = wmma_bf16(a0, b0[sub], acc[sub]);
    // phase 1: prefetch k0+64 into buffer 0, compute with buffer 1
    a0 = ldfragA_f32(xrow, k0 + 64, half);
#pragma unroll
    for (int sub = 0; sub < 4; ++sub) b0[sub] = ldfragB(wrow[sub], k0 + 64, half);
#pragma unroll
    for (int sub = 0; sub < 4; ++sub) acc[sub] = wmma_bf16(a1, b1[sub], acc[sub]);
  }
  // tail: buffer 0 holds k = E_-64; load and compute final two steps
  {
    v16bf a1 = ldfragA_f32(xrow, E_ - 32, half);
    v16bf b1[4];
#pragma unroll
    for (int sub = 0; sub < 4; ++sub) b1[sub] = ldfragB(wrow[sub], E_ - 32, half);
#pragma unroll
    for (int sub = 0; sub < 4; ++sub) acc[sub] = wmma_bf16(a0, b0[sub], acc[sub]);
#pragma unroll
    for (int sub = 0; sub < 4; ++sub) acc[sub] = wmma_bf16(a1, b1[sub], acc[sub]);
  }

#pragma unroll
  for (int sub = 0; sub < 4; ++sub) {
    const int c = c0 + sub * 16 + l16;
    const int h = c & (H_ - 1);
    const int d = c >> 3;
#pragma unroll
    for (int r = 0; r < 8; ++r) {
      const int n = n0 + 8 * half + r;
      const int b = n >> 11;           // / S_
      const int s = n & (S_ - 1);
      size_t addr;
      if (mode == 0)
        addr = ((size_t)(b * H_ + h) * S_ + s) * HD_ + d;
      else
        addr = ((size_t)(b * H_ + h) * HD_ + d) * S_ + s;
      dst[addr] = (__bf16)acc[sub][r];
    }
  }
}

// ---------------- kernel 3: Q_rel = Qp x Rh^T, f32 store (stride PP_) ---------
__global__ __launch_bounds__(256) void qrel_kernel(const __bf16* __restrict__ Qp,
                                                   const __bf16* __restrict__ Rh,
                                                   float* __restrict__ Qrel) {
  const int lane = threadIdx.x & 31;
  const int wave = threadIdx.x >> 5;
  const int l16 = lane & 15;
  const int half = lane >> 4;
  const int bh = blockIdx.y;
  const int h = bh & (H_ - 1);
  const int q0 = (blockIdx.x * 8 + wave) * 16;

  const __bf16* qrow = Qp + ((size_t)bh * S_ + q0 + l16) * HD_;
  v8f acc[3];
#pragma unroll
  for (int i = 0; i < 3; ++i) acc[i] = v8f_zero();

  // load everything up-front (only 8 fragments total), then 6 WMMAs
  v16bf a0 = ldfragA(qrow, 0, half);
  v16bf a1 = ldfragA(qrow, 32, half);
  v16bf b0[3], b1[3];
#pragma unroll
  for (int sub = 0; sub < 3; ++sub) {
    const __bf16* prow = Rh + ((size_t)h * PP_ + sub * 16 + l16) * HD_;
    b0[sub] = ldfragB(prow, 0, half);
    b1[sub] = ldfragB(prow, 32, half);
  }
#pragma unroll
  for (int sub = 0; sub < 3; ++sub) {
    acc[sub] = wmma_bf16(a0, b0[sub], acc[sub]);
    acc[sub] = wmma_bf16(a1, b1[sub], acc[sub]);
  }

#pragma unroll
  for (int sub = 0; sub < 3; ++sub) {
#pragma unroll
    for (int r = 0; r < 8; ++r) {
      const int q = q0 + 8 * half + r;
      Qrel[((size_t)bh * S_ + q) * PP_ + sub * 16 + l16] = acc[sub][r];
    }
  }
}

// ---------------- kernel 4: flash attention with relative bias ----------------
__global__ __launch_bounds__(256) void attn_kernel(const __bf16* __restrict__ Qp,
                                                   const __bf16* __restrict__ Kp,
                                                   const __bf16* __restrict__ Vpt,
                                                   const float* __restrict__ Qrel,
                                                   const int* __restrict__ rpi,
                                                   __bf16* __restrict__ Ctx) {
  __shared__ __bf16 lds[8][16 * 32];   // per-wave P tile (16 x 32)

  const int lane = threadIdx.x & 31;
  const int wave = threadIdx.x >> 5;
  const int l16 = lane & 15;
  const int half = lane >> 4;
  const int bh = blockIdx.y;
  const int b = bh >> 3;
  const int h = bh & (H_ - 1);
  const int q0 = (blockIdx.x * 8 + wave) * 16;

  // Q A-fragments (row q = q0 + l16), preloaded for both halves of head dim
  const __bf16* qrow = Qp + ((size_t)bh * S_ + q0 + l16) * HD_;
  const v16bf aq0 = ldfragA(qrow, 0, half);
  const v16bf aq1 = ldfragA(qrow, 32, half);

  // V row pointers (lane's output column), fixed across the k loop
  const __bf16* vrow[4];
#pragma unroll
  for (int sub = 0; sub < 4; ++sub)
    vrow[sub] = Vpt + ((size_t)bh * HD_ + sub * 16 + l16) * S_;

  v8f o[4];
#pragma unroll
  for (int i = 0; i < 4; ++i) o[i] = v8f_zero();
  float mrow[8], lrow[8];
#pragma unroll
  for (int r = 0; r < 8; ++r) { mrow[r] = NEG_BIG; lrow[r] = 0.0f; }

  const float* qrelb = Qrel + (size_t)bh * S_ * PP_;
  const int* rpib = rpi + (size_t)b * S_ * S_;
  __bf16* ldsw = &lds[wave][0];

  // ---- tile loader: all loads for k-tile kt are independent of compute ----
  auto load_tile = [&](int kt, v16bf& k00, v16bf& k01, v16bf& k10, v16bf& k11,
                       v16bf* vv, int* r0, int* r1) {
    const __bf16* krow0 = Kp + ((size_t)bh * S_ + kt + l16) * HD_;
    const __bf16* krow1 = krow0 + 16 * HD_;
    k00 = ldfragB(krow0, 0, half);
    k01 = ldfragB(krow0, 32, half);
    k10 = ldfragB(krow1, 0, half);
    k11 = ldfragB(krow1, 32, half);
#pragma unroll
    for (int sub = 0; sub < 4; ++sub) vv[sub] = ldfragB(vrow[sub], kt, half);
#pragma unroll
    for (int r = 0; r < 8; ++r) {
      const int q = q0 + 8 * half + r;
      const int* rrow = rpib + (size_t)q * S_ + kt;
      r0[r] = rrow[l16];
      r1[r] = rrow[16 + l16];
    }
  };

  // ---- process one 16x32 energy tile held in registers ----
  auto process = [&](v16bf k00, v16bf k01, v16bf k10, v16bf k11, const v16bf* vv,
                     const int* r0, const int* r1) {
    // dependent gathers of the relative bias (issue before the WMMAs)
    float b0[8], b1[8];
#pragma unroll
    for (int r = 0; r < 8; ++r) {
      const int q = q0 + 8 * half + r;
      const float* qr = qrelb + (size_t)q * PP_;
      b0[r] = r0[r] ? qr[r0[r]] : NEG_BIG;
      b1[r] = r1[r] ? qr[r1[r]] : NEG_BIG;
    }
    // energy tiles
    v8f e0 = v8f_zero(), e1 = v8f_zero();
    e0 = wmma_bf16(aq0, k00, e0);
    e0 = wmma_bf16(aq1, k01, e0);
    e1 = wmma_bf16(aq0, k10, e1);
    e1 = wmma_bf16(aq1, k11, e1);
    // online softmax (rows m = 8*half + r)
    float p0v[8], p1v[8], alpha[8];
#pragma unroll
    for (int r = 0; r < 8; ++r) {
      const float ev0 = (e0[r] + b0[r]) * 0.125f;   // 1/sqrt(64)
      const float ev1 = (e1[r] + b1[r]) * 0.125f;
      const float tmax = rmax16(fmaxf(ev0, ev1));
      const float mnew = fmaxf(mrow[r], tmax);
      const float a = __expf(mrow[r] - mnew);
      const float x0 = __expf(ev0 - mnew);
      const float x1 = __expf(ev1 - mnew);
      lrow[r] = lrow[r] * a + rsum16(x0 + x1);
      mrow[r] = mnew;
      alpha[r] = a;
      p0v[r] = x0;
      p1v[r] = x1;
    }
    // transpose P (C layout -> A layout) through LDS
#pragma unroll
    for (int r = 0; r < 8; ++r) {
      const int m = 8 * half + r;
      ldsw[m * 32 + l16] = (__bf16)p0v[r];
      ldsw[m * 32 + 16 + l16] = (__bf16)p1v[r];
    }
    asm volatile("s_wait_dscnt 0" ::: "memory");  // DS pipe is in-order per wave
    const v16bf aP = ldfragA(ldsw + l16 * 32, 0, half);
    // rescale O and accumulate P x V
#pragma unroll
    for (int sub = 0; sub < 4; ++sub) {
      v8f oo = o[sub];
#pragma unroll
      for (int r = 0; r < 8; ++r) oo[r] *= alpha[r];
      o[sub] = wmma_bf16(aP, vv[sub], oo);
    }
  };

  // ---- ping-pong pipelined main loop (2 tiles per iteration) ----
  v16bf ak00, ak01, ak10, ak11, av[4];
  int ar0[8], ar1[8];
  v16bf bk00, bk01, bk10, bk11, bv[4];
  int br0[8], br1[8];
  load_tile(0, ak00, ak01, ak10, ak11, av, ar0, ar1);

#pragma unroll 1
  for (int k0 = 0; k0 < S_; k0 += 64) {
    // prefetch tile k0+32 into buffer B (always in range: k0+32 <= S_-32)
    load_tile(k0 + 32, bk00, bk01, bk10, bk11, bv, br0, br1);
    process(ak00, ak01, ak10, ak11, av, ar0, ar1);
    // prefetch tile k0+64 into buffer A (clamped on the last iteration to a
    // valid tile so the loop body stays branch-free and in-bounds)
    const int kn = (k0 + 64 < S_) ? k0 + 64 : k0;
    load_tile(kn, ak00, ak01, ak10, ak11, av, ar0, ar1);
    process(bk00, bk01, bk10, bk11, bv, br0, br1);
  }

  // --- normalize and store recombined context [b][s][d*8+h] as bf16 ---
#pragma unroll
  for (int r = 0; r < 8; ++r) lrow[r] = 1.0f / lrow[r];
#pragma unroll
  for (int sub = 0; sub < 4; ++sub) {
    const int d = sub * 16 + l16;
#pragma unroll
    for (int r = 0; r < 8; ++r) {
      const int q = q0 + 8 * half + r;
      Ctx[(size_t)(b * S_ + q) * E_ + d * H_ + h] = (__bf16)(o[sub][r] * lrow[r]);
    }
  }
}

// ---------------- kernel 5: output GEMM  Out = Ctx * Wo^T (f32 store) ---------
__global__ __launch_bounds__(256) void oproj_kernel(const __bf16* __restrict__ Ctx,
                                                    const __bf16* __restrict__ Wo,
                                                    float* __restrict__ Out) {
  const int lane = threadIdx.x & 31;
  const int wave = threadIdx.x >> 5;
  const int l16 = lane & 15;
  const int half = lane >> 4;
  const int n0 = blockIdx.x * 16;
  const int c0 = wave * 64;

  const __bf16* xrow = Ctx + (size_t)(n0 + l16) * E_;
  const __bf16* wrow[4];
#pragma unroll
  for (int sub = 0; sub < 4; ++sub)
    wrow[sub] = Wo + (size_t)(c0 + sub * 16 + l16) * E_;

  v8f acc[4];
#pragma unroll
  for (int i = 0; i < 4; ++i) acc[i] = v8f_zero();

  v16bf a0 = ldfragA(xrow, 0, half);
  v16bf b0[4];
#pragma unroll
  for (int sub = 0; sub < 4; ++sub) b0[sub] = ldfragB(wrow[sub], 0, half);

#pragma unroll 1
  for (int k0 = 0; k0 < E_ - 64; k0 += 64) {
    v16bf a1 = ldfragA(xrow, k0 + 32, half);
    v16bf b1[4];
#pragma unroll
    for (int sub = 0; sub < 4; ++sub) b1[sub] = ldfragB(wrow[sub], k0 + 32, half);
#pragma unroll
    for (int sub = 0; sub < 4; ++sub) acc[sub] = wmma_bf16(a0, b0[sub], acc[sub]);
    a0 = ldfragA(xrow, k0 + 64, half);
#pragma unroll
    for (int sub = 0; sub < 4; ++sub) b0[sub] = ldfragB(wrow[sub], k0 + 64, half);
#pragma unroll
    for (int sub = 0; sub < 4; ++sub) acc[sub] = wmma_bf16(a1, b1[sub], acc[sub]);
  }
  {
    v16bf a1 = ldfragA(xrow, E_ - 32, half);
    v16bf b1[4];
#pragma unroll
    for (int sub = 0; sub < 4; ++sub) b1[sub] = ldfragB(wrow[sub], E_ - 32, half);
#pragma unroll
    for (int sub = 0; sub < 4; ++sub) acc[sub] = wmma_bf16(a0, b0[sub], acc[sub]);
#pragma unroll
    for (int sub = 0; sub < 4; ++sub) acc[sub] = wmma_bf16(a1, b1[sub], acc[sub]);
  }

#pragma unroll
  for (int sub = 0; sub < 4; ++sub) {
#pragma unroll
    for (int r = 0; r < 8; ++r) {
      Out[(size_t)(n0 + 8 * half + r) * E_ + c0 + sub * 16 + l16] = acc[sub][r];
    }
  }
}

// ---------------- launch ----------------
extern "C" void kernel_launch(void* const* d_in, const int* in_sizes, int n_in,
                              void* d_out, int out_size, void* d_ws, size_t ws_size,
                              hipStream_t stream) {
  const float* Q = (const float*)d_in[0];
  const float* K = (const float*)d_in[1];
  const float* V = (const float*)d_in[2];
  const int* rpi = (const int*)d_in[3];
  const float* Wq = (const float*)d_in[4];
  const float* Wk = (const float*)d_in[5];
  const float* Wv = (const float*)d_in[6];
  const float* Wo = (const float*)d_in[7];
  const float* rel = (const float*)d_in[8];

  char* w = (char*)d_ws;
  __bf16* Wbf = (__bf16*)w;          w += (size_t)4 * E_ * E_ * 2;       // 2 MB
  __bf16* Rh = (__bf16*)w;           w += (size_t)H_ * PP_ * HD_ * 2;    // 48 KB
  __bf16* Qp = (__bf16*)w;           w += (size_t)B_ * H_ * S_ * HD_ * 2; // 4 MB
  __bf16* Kp = (__bf16*)w;           w += (size_t)B_ * H_ * S_ * HD_ * 2; // 4 MB
  __bf16* Vpt = (__bf16*)w;          w += (size_t)B_ * H_ * S_ * HD_ * 2; // 4 MB
  float* Qrel = (float*)w;           w += (size_t)B_ * H_ * S_ * PP_ * 4; // 6 MB
  __bf16* Ctx = (__bf16*)w;          w += (size_t)B_ * S_ * E_ * 2;       // 4 MB

  // 1) bf16 conversion of weights + per-head rel embedding table
  {
    const int total = 4 * E_ * E_ + H_ * PP_ * HD_;
    cvt_kernel<<<(total + 255) / 256, 256, 0, stream>>>(Wq, Wk, Wv, Wo, rel, Wbf, Rh);
  }
  // 2) projections (head-split layouts)
  proj_kernel<<<dim3((B_ * S_) / 16), 256, 0, stream>>>(Q, Wbf + 0 * E_ * E_, Qp, 0);
  proj_kernel<<<dim3((B_ * S_) / 16), 256, 0, stream>>>(K, Wbf + 1 * E_ * E_, Kp, 0);
  proj_kernel<<<dim3((B_ * S_) / 16), 256, 0, stream>>>(V, Wbf + 2 * E_ * E_, Vpt, 1);
  // 3) Q_rel
  qrel_kernel<<<dim3(S_ / 16 / 8, B_ * H_), 256, 0, stream>>>(Qp, Rh, Qrel);
  // 4) flash attention with relative bias
  attn_kernel<<<dim3(S_ / 16 / 8, B_ * H_), 256, 0, stream>>>(Qp, Kp, Vpt, Qrel, rpi,
                                                              Ctx);
  // 5) output projection
  oproj_kernel<<<dim3((B_ * S_) / 16), 256, 0, stream>>>(Ctx, Wbf + 3 * E_ * E_,
                                                         (float*)d_out);
}
